// GCN_Net1_43052752175665
// MI455X (gfx1250) — matrix operand order repaired and checked by reading, not compiled, not run on hardware
//
#include <hip/hip_runtime.h>
#include <hip/hip_bf16.h>

typedef __attribute__((ext_vector_type(16))) __bf16 v16bf;
typedef __attribute__((ext_vector_type(8)))  float  v8f;

#define N_NODES 50000
#define N_EDGES 800000
#define F_IN    512
#define F_HID   64
#define F_OUT   16
#define M_TILES (N_NODES / 16)   // 3125, exact

__device__ __forceinline__ void atomAddF32(float* p, float v) {
    __hip_atomic_fetch_add(p, v, __ATOMIC_RELAXED, __HIP_MEMORY_SCOPE_AGENT);
}

// ---------------- degree / normalization ----------------
__global__ void k_init_deg(float* deg) {
    int i = blockIdx.x * blockDim.x + threadIdx.x;
    if (i < N_NODES) deg[i] = 1.0f;            // self-loop
}
__global__ void k_accum_deg(const long long* __restrict__ ei, float* deg) {
    int e = blockIdx.x * blockDim.x + threadIdx.x;
    if (e < N_EDGES) atomAddF32(&deg[(int)ei[N_EDGES + e]], 1.0f);
}
__global__ void k_finish_dinv(float* deg) {
    int i = blockIdx.x * blockDim.x + threadIdx.x;
    if (i < N_NODES) { float v = deg[i]; deg[i] = (v > 0.f) ? rsqrtf(v) : 0.f; }
}

// ---------------- weight fragmentization (B-matrix VGPR layout) ----------------
// B 32x16 (KxN) per 16x16x32 step: lane group g=lane>>4 covers K = g*8 + (i&7) + 16*(i>>3), N = lane&15.
__global__ void k_prep_w1(const float* __restrict__ W1, __bf16* __restrict__ w1f) {
    int t = blockIdx.x * blockDim.x + threadIdx.x;           // 16*4*32*16 = 32768
    if (t >= (F_IN / 32) * 4 * 32 * 16) return;
    int i = t & 15, lane = (t >> 4) & 31, nt = (t >> 9) & 3, kb = t >> 11;
    int k = kb * 32 + ((lane >> 4) << 3) + (i & 7) + ((i >> 3) << 4);
    int n = nt * 16 + (lane & 15);
    w1f[t] = (__bf16)W1[k * F_HID + n];
}
__global__ void k_prep_w2(const float* __restrict__ W2, __bf16* __restrict__ w2f) {
    int t = blockIdx.x * blockDim.x + threadIdx.x;           // 2*32*16 = 1024
    if (t >= (F_HID / 32) * 32 * 16) return;
    int i = t & 15, lane = (t >> 4) & 31, kb = t >> 9;
    int k = kb * 32 + ((lane >> 4) << 3) + (i & 7) + ((i >> 3) << 4);
    int n = lane & 15;
    w2f[t] = (__bf16)W2[k * F_OUT + n];
}

// ---------------- GEMM1: h1 = x @ W1  (bf16 WMMA, f32 accumulate) ----------------
__global__ __launch_bounds__(256) void k_gemm1(const float* __restrict__ x,
                                               const __bf16* __restrict__ w1f,
                                               float* __restrict__ h1) {
    const int wave = blockIdx.x * 8 + (threadIdx.x >> 5);
    const int lane = threadIdx.x & 31;
    if (wave >= M_TILES) return;                              // wave-uniform: EXEC stays all-1s
    const int m  = wave * 16 + (lane & 15);
    const int kg = (lane >> 4) * 8;                           // A layout: k = 8g + 8*(i>>3) + i
    const float* __restrict__ xrow = x + (size_t)m * F_IN;
    const v16bf* __restrict__ wb = ((const v16bf*)w1f) + lane;
    v8f c0 = {}, c1 = {}, c2 = {}, c3 = {};
    for (int kb = 0; kb < F_IN / 32; ++kb) {
        const float4 f0 = *(const float4*)(xrow + kb * 32 + kg);
        const float4 f1 = *(const float4*)(xrow + kb * 32 + kg + 4);
        const float4 f2 = *(const float4*)(xrow + kb * 32 + kg + 16);
        const float4 f3 = *(const float4*)(xrow + kb * 32 + kg + 20);
        v16bf a;
        a[0]=(__bf16)f0.x;  a[1]=(__bf16)f0.y;  a[2]=(__bf16)f0.z;  a[3]=(__bf16)f0.w;
        a[4]=(__bf16)f1.x;  a[5]=(__bf16)f1.y;  a[6]=(__bf16)f1.z;  a[7]=(__bf16)f1.w;
        a[8]=(__bf16)f2.x;  a[9]=(__bf16)f2.y;  a[10]=(__bf16)f2.z; a[11]=(__bf16)f2.w;
        a[12]=(__bf16)f3.x; a[13]=(__bf16)f3.y; a[14]=(__bf16)f3.z; a[15]=(__bf16)f3.w;
        const v16bf b0 = wb[kb * 128 +  0];
        const v16bf b1 = wb[kb * 128 + 32];
        const v16bf b2 = wb[kb * 128 + 64];
        const v16bf b3 = wb[kb * 128 + 96];
        c0 = __builtin_amdgcn_wmma_f32_16x16x32_bf16(false, a, false, b0, (short)0, c0, false, false);
        c1 = __builtin_amdgcn_wmma_f32_16x16x32_bf16(false, a, false, b1, (short)0, c1, false, false);
        c2 = __builtin_amdgcn_wmma_f32_16x16x32_bf16(false, a, false, b2, (short)0, c2, false, false);
        c3 = __builtin_amdgcn_wmma_f32_16x16x32_bf16(false, a, false, b3, (short)0, c3, false, false);
    }
    const int row0 = wave * 16 + (lane >> 4) * 8;             // C layout: M = r + 8g, N = lane&15
    const int col  = lane & 15;
#pragma unroll
    for (int r = 0; r < 8; ++r) {
        float* o = h1 + (size_t)(row0 + r) * F_HID + col;
        o[0]  = c0[r];
        o[16] = c1[r];
        o[32] = c2[r];
        o[48] = c3[r];
    }
}

// ---------------- GEMM2: h2 = relu(agg1) @ W2 (ReLU fused into A-load) ----------------
__global__ __launch_bounds__(256) void k_gemm2(const float* __restrict__ h,
                                               const __bf16* __restrict__ w2f,
                                               float* __restrict__ h2) {
    const int wave = blockIdx.x * 8 + (threadIdx.x >> 5);
    const int lane = threadIdx.x & 31;
    if (wave >= M_TILES) return;
    const int m  = wave * 16 + (lane & 15);
    const int kg = (lane >> 4) * 8;
    const float* __restrict__ hrow = h + (size_t)m * F_HID;
    const v16bf* __restrict__ wb = ((const v16bf*)w2f) + lane;
    v8f c = {};
#pragma unroll
    for (int kb = 0; kb < F_HID / 32; ++kb) {
        const float4 f0 = *(const float4*)(hrow + kb * 32 + kg);
        const float4 f1 = *(const float4*)(hrow + kb * 32 + kg + 4);
        const float4 f2 = *(const float4*)(hrow + kb * 32 + kg + 16);
        const float4 f3 = *(const float4*)(hrow + kb * 32 + kg + 20);
        v16bf a;
        a[0]=(__bf16)fmaxf(f0.x,0.f);  a[1]=(__bf16)fmaxf(f0.y,0.f);  a[2]=(__bf16)fmaxf(f0.z,0.f);  a[3]=(__bf16)fmaxf(f0.w,0.f);
        a[4]=(__bf16)fmaxf(f1.x,0.f);  a[5]=(__bf16)fmaxf(f1.y,0.f);  a[6]=(__bf16)fmaxf(f1.z,0.f);  a[7]=(__bf16)fmaxf(f1.w,0.f);
        a[8]=(__bf16)fmaxf(f2.x,0.f);  a[9]=(__bf16)fmaxf(f2.y,0.f);  a[10]=(__bf16)fmaxf(f2.z,0.f); a[11]=(__bf16)fmaxf(f2.w,0.f);
        a[12]=(__bf16)fmaxf(f3.x,0.f); a[13]=(__bf16)fmaxf(f3.y,0.f); a[14]=(__bf16)fmaxf(f3.z,0.f); a[15]=(__bf16)fmaxf(f3.w,0.f);
        const v16bf b = wb[kb * 32];
        c = __builtin_amdgcn_wmma_f32_16x16x32_bf16(false, a, false, b, (short)0, c, false, false);
    }
    const int row0 = wave * 16 + (lane >> 4) * 8;
    const int col  = lane & 15;
#pragma unroll
    for (int r = 0; r < 8; ++r)
        h2[(size_t)(row0 + r) * F_OUT + col] = c[r];
}

// ---------------- aggregation (self-loop seed + edge scatter-add) ----------------
__global__ void k_self1(const float* __restrict__ h1, const float* __restrict__ dinv,
                        const float* __restrict__ b1, float* __restrict__ out1) {
    int t = blockIdx.x * blockDim.x + threadIdx.x;
    if (t < N_NODES * F_HID) {
        int i = t >> 6, f = t & 63;
        float di = dinv[i];
        out1[t] = h1[t] * di * di + b1[f];
    }
}
__global__ void k_edge1(const long long* __restrict__ ei, const float* __restrict__ dinv,
                        const float* __restrict__ h1, float* __restrict__ out1) {
    int t = blockIdx.x * blockDim.x + threadIdx.x;
    if (t >= N_EDGES * 4) return;
    int e = t >> 2, coff = (t & 3) * 16;
    int s = (int)ei[e], d = (int)ei[N_EDGES + e];
    float norm = dinv[s] * dinv[d];
    const float4* hp = (const float4*)(h1 + (size_t)s * F_HID + coff);
    float* op = out1 + (size_t)d * F_HID + coff;
#pragma unroll
    for (int q = 0; q < 4; ++q) {
        float4 v = hp[q];
        atomAddF32(op + q * 4 + 0, v.x * norm);
        atomAddF32(op + q * 4 + 1, v.y * norm);
        atomAddF32(op + q * 4 + 2, v.z * norm);
        atomAddF32(op + q * 4 + 3, v.w * norm);
    }
}
__global__ void k_self2(const float* __restrict__ h2, const float* __restrict__ dinv,
                        const float* __restrict__ b2, float* __restrict__ out) {
    int t = blockIdx.x * blockDim.x + threadIdx.x;
    if (t < N_NODES * F_OUT) {
        int i = t >> 4, f = t & 15;
        float di = dinv[i];
        out[t] = h2[t] * di * di + b2[f];
    }
}
__global__ void k_edge2(const long long* __restrict__ ei, const float* __restrict__ dinv,
                        const float* __restrict__ h2, float* __restrict__ out) {
    int e = blockIdx.x * blockDim.x + threadIdx.x;
    if (e >= N_EDGES) return;
    int s = (int)ei[e], d = (int)ei[N_EDGES + e];
    float norm = dinv[s] * dinv[d];
    const float4* hp = (const float4*)(h2 + (size_t)s * F_OUT);
    float* op = out + (size_t)d * F_OUT;
#pragma unroll
    for (int q = 0; q < 4; ++q) {
        float4 v = hp[q];
        atomAddF32(op + q * 4 + 0, v.x * norm);
        atomAddF32(op + q * 4 + 1, v.y * norm);
        atomAddF32(op + q * 4 + 2, v.z * norm);
        atomAddF32(op + q * 4 + 3, v.w * norm);
    }
}

extern "C" void kernel_launch(void* const* d_in, const int* in_sizes, int n_in,
                              void* d_out, int out_size, void* d_ws, size_t ws_size,
                              hipStream_t stream) {
    const float*     x  = (const float*)d_in[0];
    const long long* ei = (const long long*)d_in[1];
    const float*     W1 = (const float*)d_in[2];
    const float*     b1 = (const float*)d_in[3];
    const float*     W2 = (const float*)d_in[4];
    const float*     b2 = (const float*)d_in[5];
    float*           out = (float*)d_out;

    // workspace carve-up (256B aligned)
    char* ws = (char*)d_ws;
    size_t off = 0;
    auto carve = [&](size_t bytes) { void* p = ws + off; off = (off + bytes + 255) & ~(size_t)255; return p; };
    float*  dinv = (float*)carve(N_NODES * sizeof(float));
    __bf16* w1f  = (__bf16*)carve((size_t)(F_IN / 32) * 4 * 32 * 16 * sizeof(__bf16));
    __bf16* w2f  = (__bf16*)carve((size_t)(F_HID / 32) * 32 * 16 * sizeof(__bf16));
    float*  h1   = (float*)carve((size_t)N_NODES * F_HID * sizeof(float));
    float*  o1   = (float*)carve((size_t)N_NODES * F_HID * sizeof(float));
    float*  h2   = (float*)carve((size_t)N_NODES * F_OUT * sizeof(float));
    (void)ws_size; (void)n_in; (void)in_sizes; (void)out_size;

    const int B = 256;
    k_init_deg   <<<(N_NODES + B - 1) / B, B, 0, stream>>>(dinv);
    k_accum_deg  <<<(N_EDGES + B - 1) / B, B, 0, stream>>>(ei, dinv);
    k_finish_dinv<<<(N_NODES + B - 1) / B, B, 0, stream>>>(dinv);
    k_prep_w1    <<<(32768 + B - 1) / B, B, 0, stream>>>(W1, w1f);
    k_prep_w2    <<<(1024 + B - 1) / B, B, 0, stream>>>(W2, w2f);

    k_gemm1<<<(M_TILES + 7) / 8, B, 0, stream>>>(x, w1f, h1);
    k_self1<<<(N_NODES * F_HID + B - 1) / B, B, 0, stream>>>(h1, dinv, b1, o1);
    k_edge1<<<(N_EDGES * 4 + B - 1) / B, B, 0, stream>>>(ei, dinv, h1, o1);

    k_gemm2<<<(M_TILES + 7) / 8, B, 0, stream>>>(o1, w2f, h2);
    k_self2<<<(N_NODES * F_OUT + B - 1) / B, B, 0, stream>>>(h2, dinv, b2, out);
    k_edge2<<<(N_EDGES + B - 1) / B, B, 0, stream>>>(ei, dinv, h2, out);
}
